// RecoNet_17480516895188
// MI455X (gfx1250) — compile-verified, matrix operand order checked by reading, not run on hardware
//
#include <hip/hip_runtime.h>

// RecoNet fused pipeline for MI455X (gfx1250, wave32, WMMA).
// Shapes: B=16, C=64, H=192, W=192, R=64, HW=36864.

#define BDIM 16
#define CDIM 64
#define HDIM 192
#define WDIM 192
#define RDIM 64
#define HW_  (HDIM * WDIM)

typedef __attribute__((ext_vector_type(16))) _Float16 v16h;
typedef __attribute__((ext_vector_type(8)))  float    v8f;
typedef __attribute__((ext_vector_type(4)))  float    v4f;

// ---------------------------------------------------------------- kernel 0
__global__ void k_zero(float* __restrict__ p, int n) {
    int i = blockIdx.x * 256 + threadIdx.x;
    if (i < n) p[i] = 0.0f;
}

// ---------------------------------------------------------------- kernel 1
// One block per (b,c) plane (1024 blocks, 192 threads).
// Pass 1: column sums (coalesced) -> wp_sum + plane total -> cp.
// Pass 2: row sums (plane now L2-resident) -> hp_sum.
__global__ void k_pool(const float* __restrict__ X,
                       float* __restrict__ hp_sum,   // [B][H] raw sums over (c,w)
                       float* __restrict__ wp_sum,   // [B][W] raw sums over (c,h)
                       float* __restrict__ cp) {     // [B][C] means
    __shared__ float sh[WDIM];
    const int bc = blockIdx.x;
    const int b  = bc >> 6;
    const int c  = bc & 63;
    const float* plane = X + (size_t)bc * HW_;
    const int t = threadIdx.x;              // 0..191

    float col = 0.0f;
    for (int h = 0; h < HDIM; ++h) col += plane[h * WDIM + t];
    atomicAdd(&wp_sum[b * WDIM + t], col);
    sh[t] = col;
    __syncthreads();
    if (t == 0) {
        float tot = 0.0f;
        for (int i = 0; i < WDIM; ++i) tot += sh[i];
        cp[b * CDIM + c] = tot * (1.0f / (float)HW_);
    }
    float row = 0.0f;
    const float* rp = plane + t * WDIM;
    for (int w = 0; w < WDIM; ++w) row += rp[w];
    atomicAdd(&hp_sum[b * HDIM + t], row);
}

// ---------------------------------------------------------------- kernel 2
// out[b][d][r] = (mulP ? p[r] : 1) * sigmoid( dot(Wmat[r][d][:], pool[b][:]*scale) + bias[r][d] )
__global__ void k_affine(const float* __restrict__ Wmat,
                         const float* __restrict__ bias,
                         const float* __restrict__ pool,
                         const float* __restrict__ pvec,
                         float* __restrict__ out,
                         int D, int K, float scale, int mulP) {
    extern __shared__ float pl[];           // [K][16]
    const int r = blockIdx.x;
    const int t = threadIdx.x;
    for (int i = t; i < BDIM * K; i += 256) {
        int b = i / K, k = i % K;
        pl[k * BDIM + b] = pool[i] * scale;
    }
    __syncthreads();
    const int   b  = t & 15;
    const int   dg = t >> 4;                // 0..15
    const float pr = mulP ? pvec[r] : 1.0f;
    for (int d = dg; d < D; d += 16) {
        const float* wr = Wmat + ((size_t)r * D + d) * K;
        float acc = bias[r * D + d];
        for (int k = 0; k < K; ++k) acc += wr[k] * pl[k * BDIM + b];
        float s = 1.0f / (1.0f + __expf(-acc));
        out[((size_t)b * D + d) * RDIM + r] = s * pr;
    }
}

// ---------------------------------------------------------------- kernel 3
// Fused: y[b,c,pos] = sum_r A[b][c][r] * (Hw[b][h][r] * Ww[b][w][r]);  out = y * X
// grid (16, 36), 256 threads (8 waves). LDS: A(16KB) + Hw(48KB) + Ww(48KB) = 112KB.
// WMMA f32 = f16 x f16: M=c(4 tiles of 16), N=16 hw positions (fixed h), K=r (2 steps of 32).
// 16-bit fragment K-layout per lane group g: K in [8g,8g+8) U [16+8g,16+8g+8)  (+32*ks).

__device__ __forceinline__ v16h mk_frag(v4f q0, v4f q1, v4f q2, v4f q3) {
    v16h r;
#pragma unroll
    for (int e = 0; e < 4; ++e) {
        r[e]      = (_Float16)q0[e];
        r[4 + e]  = (_Float16)q1[e];
        r[8 + e]  = (_Float16)q2[e];
        r[12 + e] = (_Float16)q3[e];
    }
    return r;
}

__global__ void __launch_bounds__(256)
k_main(const float* __restrict__ X,
       const float* __restrict__ Ag,   // [B][C][R]
       const float* __restrict__ Hg,   // [B][H][R]
       const float* __restrict__ Wg,   // [B][W][R]
       float* __restrict__ out) {
    extern __shared__ float lds[];
    float* lA = lds;                    // 64*64
    float* lH = lds + 4096;             // 192*64
    float* lW = lds + 4096 + 12288;     // 192*64
    const int b     = blockIdx.x;
    const int chunk = blockIdx.y;       // 0..35, 64 n-tiles each
    const int t     = threadIdx.x;

    {   // stage per-b tables (contiguous, float4-coalesced)
        const float4* a4 = (const float4*)(Ag + (size_t)b * 4096);
        const float4* h4 = (const float4*)(Hg + (size_t)b * 12288);
        const float4* w4 = (const float4*)(Wg + (size_t)b * 12288);
        float4* la4 = (float4*)lA; float4* lh4 = (float4*)lH; float4* lw4 = (float4*)lW;
        for (int i = t; i < 1024; i += 256) la4[i] = a4[i];
        for (int i = t; i < 3072; i += 256) lh4[i] = h4[i];
        for (int i = t; i < 3072; i += 256) lw4[i] = w4[i];
    }
    __syncthreads();

    const int wave = t >> 5;
    const int lane = t & 31;
    const int g    = lane >> 4;         // lane group
    const int mn   = lane & 15;         // M (for A) / N (for B,C,D)

    // Build the 8 A fragments once per wave (batched ds_load_b128, then cvt).
    v16h afrag[8];
#pragma unroll
    for (int mt = 0; mt < 4; ++mt) {
        const v4f* arow4 = (const v4f*)(lA + (mt * 16 + mn) * 64);
#pragma unroll
        for (int ks = 0; ks < 2; ++ks) {
            v4f a0 = arow4[ks * 8 + 2 * g];
            v4f a1 = arow4[ks * 8 + 2 * g + 1];
            v4f a2 = arow4[ks * 8 + 4 + 2 * g];
            v4f a3 = arow4[ks * 8 + 4 + 2 * g + 1];
            afrag[mt * 2 + ks] = mk_frag(a0, a1, a2, a3);
        }
    }

    // 8 n-tiles per wave; each tile = 64(M) x 16(N) outputs.
    for (int i = 0; i < 8; ++i) {
        const int tile = chunk * 64 + i * 8 + wave;     // 0..2303
        const int h    = tile / 12;                     // 192/16 = 12 tiles per row
        const int w0   = (tile % 12) * 16;
        const int pos0 = h * WDIM + w0;
        const v4f* h4 = (const v4f*)(lH + h * 64);
        const v4f* w4 = (const v4f*)(lW + (w0 + mn) * 64);

        // Prefetch X epilogue operands: latency hidden by fragment build + WMMAs.
        const size_t base = ((size_t)(b * CDIM)) * HW_ + pos0 + mn;
        float xv[32];
#pragma unroll
        for (int mt = 0; mt < 4; ++mt)
#pragma unroll
            for (int d = 0; d < 8; ++d)
                xv[mt * 8 + d] = X[base + (size_t)(mt * 16 + d + 8 * g) * HW_];

        v8f acc[4] = {};
#pragma unroll
        for (int ks = 0; ks < 2; ++ks) {
            // 8 x ds_load_b128, one wait, then pk_mul + pk_cvt.
            v4f h0 = h4[ks * 8 + 2 * g];
            v4f h1 = h4[ks * 8 + 2 * g + 1];
            v4f h2 = h4[ks * 8 + 4 + 2 * g];
            v4f h3 = h4[ks * 8 + 4 + 2 * g + 1];
            v4f q0 = w4[ks * 8 + 2 * g];
            v4f q1 = w4[ks * 8 + 2 * g + 1];
            v4f q2 = w4[ks * 8 + 4 + 2 * g];
            v4f q3 = w4[ks * 8 + 4 + 2 * g + 1];
            v16h bfr = mk_frag(h0 * q0, h1 * q1, h2 * q2, h3 * q3);
#pragma unroll
            for (int mt = 0; mt < 4; ++mt)
                acc[mt] = __builtin_amdgcn_wmma_f32_16x16x32_f16(
                    false, afrag[mt * 2 + ks], false, bfr,
                    (short)0, acc[mt], false, false);
        }

        // Fused epilogue: out = D * X.  D layout: m = mt*16 + d + 8g, n = mn.
#pragma unroll
        for (int mt = 0; mt < 4; ++mt)
#pragma unroll
            for (int d = 0; d < 8; ++d) {
                const size_t idx = base + (size_t)(mt * 16 + d + 8 * g) * HW_;
                out[idx] = acc[mt][d] * xv[mt * 8 + d];
            }
    }
}

// ---------------------------------------------------------------- launch
extern "C" void kernel_launch(void* const* d_in, const int* in_sizes, int n_in,
                              void* d_out, int out_size, void* d_ws, size_t ws_size,
                              hipStream_t stream) {
    const float* X  = (const float*)d_in[0];
    const float* p  = (const float*)d_in[1];
    const float* Wc = (const float*)d_in[2];
    const float* bc = (const float*)d_in[3];
    const float* Wh = (const float*)d_in[4];
    const float* bh = (const float*)d_in[5];
    const float* Ww = (const float*)d_in[6];
    const float* bw = (const float*)d_in[7];
    float* out = (float*)d_out;

    float* ws     = (float*)d_ws;
    float* hp_sum = ws;                 //  3072   [B][H]
    float* wp_sum = ws + 3072;          //  3072   [B][W]
    float* cp     = ws + 6144;          //  1024   [B][C]
    float* Ag     = ws + 7168;          //  65536  [B][C][R]
    float* Hg     = ws + 72704;         // 196608  [B][H][R]
    float* Wg     = ws + 269312;        // 196608  [B][W][R]

    (void)in_sizes; (void)n_in; (void)out_size; (void)ws_size;

    hipFuncSetAttribute((const void*)k_main,
                        hipFuncAttributeMaxDynamicSharedMemorySize, 114688);

    hipLaunchKernelGGL(k_zero, dim3(24), dim3(256), 0, stream, hp_sum, 6144);
    hipLaunchKernelGGL(k_pool, dim3(BDIM * CDIM), dim3(HDIM), 0, stream,
                       X, hp_sum, wp_sum, cp);
    const float invCW = 1.0f / (float)(CDIM * WDIM);   // = 1/(C*H) too
    hipLaunchKernelGGL(k_affine, dim3(RDIM), dim3(256), CDIM * BDIM * 4, stream,
                       Wc, bc, cp, p, Ag, CDIM, CDIM, 1.0f, 1);
    hipLaunchKernelGGL(k_affine, dim3(RDIM), dim3(256), HDIM * BDIM * 4, stream,
                       Wh, bh, hp_sum, p, Hg, HDIM, HDIM, invCW, 0);
    hipLaunchKernelGGL(k_affine, dim3(RDIM), dim3(256), WDIM * BDIM * 4, stream,
                       Ww, bw, wp_sum, p, Wg, WDIM, WDIM, invCW, 0);
    hipLaunchKernelGGL(k_main, dim3(BDIM, 36), dim3(256), 114688, stream,
                       X, Ag, Hg, Wg, out);
}
